// SpatiotemporalGraphTransformer_84490596647232
// MI455X (gfx1250) — compile-verified, matrix operand order checked by reading, not run on hardware
//
#include <hip/hip_runtime.h>
#include <hip/hip_bf16.h>

// ---------------------------------------------------------------------------
// Problem constants (from the reference)
// ---------------------------------------------------------------------------
#define BT   240          // T
#define NN   32           // N persons
#define TOK  7680         // T*N tokens
#define DIN  256
#define DD   512
#define LL   2
#define HH   8
#define HD   64
#define TWIN 10           // TW
#define CSZ  30           // CS
#define CC   8            // C = T/CS
#define WTK  50           // WT = CS + 2*TW
#define QN   960          // CS*N queries per chunk
#define KN   1600         // WT*N keys per chunk

typedef __attribute__((ext_vector_type(16))) _Float16 v16h;
typedef __attribute__((ext_vector_type(8)))  float    v8f;

static __device__ __forceinline__ v8f wmma_f16(v16h a, v16h b, v8f c) {
  // D = A(16x32 f16) * B(32x16 f16) + C(16x16 f32)
  return __builtin_amdgcn_wmma_f32_16x16x32_f16(
      /*neg_a=*/false, a, /*neg_b=*/false, b,
      /*c_mod=*/(short)0, c, /*reuse_a=*/false, /*reuse_b=*/false);
}

// ---------------------------------------------------------------------------
// Generic WMMA GEMM:  C[M,N] = act( A[M,K] * W[K,N] + bias[N] )
// Compile-time N/K/ACT so weight-transpose staging uses one base address +
// constant immediate offsets (clauseable global_load burst, no addr chains).
// Block = 256 threads = 8 waves; block tile 64x128; wave tile 32x32; K step 32.
// A staged row-major [row][k]; B staged TRANSPOSED [col][k] so that both A and
// B fragments are contiguous 8-half runs -> ds_load_b128 pairs.
// ---------------------------------------------------------------------------
template <int N, int K, int ACT>
__global__ __launch_bounds__(256)
void gemm_wmma(const float* __restrict__ A, const float* __restrict__ W,
               const float* __restrict__ bias, float* __restrict__ C, int M) {
  const int tid  = threadIdx.x;
  const int lane = tid & 31;
  const int wave = tid >> 5;
  const int wm   = wave >> 2;      // 0..1
  const int wn   = wave & 3;       // 0..3
  const int g    = lane >> 4;      // lane group
  const int lr   = lane & 15;

  const int m0 = blockIdx.y * 64;
  const int n0 = blockIdx.x * 128;

  __shared__ _Float16 As[64 * 40];    // [row][k], stride 40 halves (80B, 16B-aligned)
  __shared__ _Float16 Bs[128 * 40];   // [col][k] TRANSPOSED, stride 40 halves

  v8f acc[2][2] = {};

  // A staging coords: 8 contiguous elems/thread
  const int arow = tid >> 2;          // 0..63
  const int acb  = (tid & 3) * 8;     // 0,8,16,24
  const float* asrc = A + (size_t)(m0 + arow) * K + acb;

  // B staging coords: each thread owns one column half (16 k-values)
  const int bcol  = tid & 127;        // 0..127
  const int bkoff = (tid >> 7) * 16;  // 0 or 16
  const float* wsrc = W + (size_t)bkoff * N + n0 + bcol;

  for (int k0 = 0; k0 < K; k0 += 32) {
    // --- stage A tile (64x32): two float4 loads ---
    {
      const float4 a4a = ((const float4*)(asrc + k0))[0];
      const float4 a4b = ((const float4*)(asrc + k0))[1];
      _Float16* dst = As + arow * 40 + acb;
      dst[0] = (_Float16)a4a.x; dst[1] = (_Float16)a4a.y;
      dst[2] = (_Float16)a4a.z; dst[3] = (_Float16)a4a.w;
      dst[4] = (_Float16)a4b.x; dst[5] = (_Float16)a4b.y;
      dst[6] = (_Float16)a4b.z; dst[7] = (_Float16)a4b.w;
    }
    // --- stage B tile (32x128) transposed: base + constant offsets ---
    {
      const float* src = wsrc + (size_t)k0 * N;
      _Float16 tmp[16];
#pragma unroll
      for (int j = 0; j < 16; ++j) tmp[j] = (_Float16)src[j * N];
      _Float16* dst = Bs + bcol * 40 + bkoff;
#pragma unroll
      for (int j = 0; j < 16; ++j) dst[j] = tmp[j];
      if (k0 + 32 < K)   // prefetch next K-slice of weights (global_prefetch_b8)
        __builtin_prefetch(src + 32 * N, 0, 1);
    }
    __syncthreads();

    // --- load fragments (all contiguous -> ds_load_b128 pairs), 4 WMMAs ---
    v16h a0, a1, b0, b1;
#pragma unroll
    for (int i = 0; i < 8; ++i) {
      a0[i]     = As[(wm * 32 + lr) * 40 + g * 8 + i];
      a0[i + 8] = As[(wm * 32 + lr) * 40 + 16 + g * 8 + i];
      a1[i]     = As[(wm * 32 + 16 + lr) * 40 + g * 8 + i];
      a1[i + 8] = As[(wm * 32 + 16 + lr) * 40 + 16 + g * 8 + i];
      b0[i]     = Bs[(wn * 32 + lr) * 40 + g * 8 + i];
      b0[i + 8] = Bs[(wn * 32 + lr) * 40 + 16 + g * 8 + i];
      b1[i]     = Bs[(wn * 32 + 16 + lr) * 40 + g * 8 + i];
      b1[i + 8] = Bs[(wn * 32 + 16 + lr) * 40 + 16 + g * 8 + i];
    }
    acc[0][0] = wmma_f16(a0, b0, acc[0][0]);
    acc[0][1] = wmma_f16(a0, b1, acc[0][1]);
    acc[1][0] = wmma_f16(a1, b0, acc[1][0]);
    acc[1][1] = wmma_f16(a1, b1, acc[1][1]);
    __syncthreads();
  }

  // --- epilogue: bias (+ GELU) + store ---
#pragma unroll
  for (int cm = 0; cm < 2; ++cm) {
#pragma unroll
    for (int cn = 0; cn < 2; ++cn) {
      const int col = n0 + wn * 32 + cn * 16 + lr;
      const float bv = bias[col];
#pragma unroll
      for (int r = 0; r < 8; ++r) {
        const int row = m0 + wm * 32 + cm * 16 + r + 8 * g;
        float v = acc[cm][cn][r] + bv;
        if (ACT == 1) {
          const float x = v;
          v = 0.5f * x * (1.0f + tanhf(0.7978845608028654f *
                                        (x + 0.044715f * x * x * x)));
        }
        C[(size_t)row * N + col] = v;
      }
    }
  }
}

// ---------------------------------------------------------------------------
// Windowed flash attention. One wave per (chunk c, head h, 16-query tile).
// qkv: [TOK][3*DD] rows = token, layout [q(512) | k(512) | v(512)], per-head 64.
// K/V staged per 32-key tile through LDS (float4 global loads, branchless
// validity): Kls[key][d] row-major, Vt[d][key] transposed -> every fragment
// load is a contiguous b128 pair.
// out: [TOK][DD] (pre-projection attention output)
// ---------------------------------------------------------------------------
__global__ __launch_bounds__(32)
void attn_wmma(const float* __restrict__ qkv, const unsigned char* __restrict__ pm,
               float* __restrict__ out) {
  const int blk  = blockIdx.x;                 // CC*HH*(QN/16) = 3840
  const int qt   = blk % (QN / 16);            // 0..59
  const int head = (blk / (QN / 16)) % HH;     // 0..7
  const int c    = blk / ((QN / 16) * HH);     // 0..7
  const int lane = threadIdx.x;
  const int g    = lane >> 4;
  const int lr   = lane & 15;
  const int qbase = qt * 16;
  const float NEGINF = -__builtin_inff();

  __shared__ _Float16 Kls[32 * 72];            // [key][d], stride 72 halves (144B)
  __shared__ _Float16 Vt[64 * 40];             // [d][key], stride 40 halves (80B)
  __shared__ _Float16 Pls[16][40];             // P tile transpose staging

  // ---- Q fragments (16 rows x 64 d), 1/sqrt(HD)=0.125 folded in ----
  v16h aq0, aq1;
  {
    const int qtok = c * QN + qbase + lr;      // both lane groups: same rows
    const float4* qp4 = (const float4*)(qkv + (size_t)qtok * (3 * DD) + head * HD);
    float qbuf[32];
    float4* qb4 = (float4*)qbuf;
    qb4[0] = qp4[g * 2 + 0];  qb4[1] = qp4[g * 2 + 1];   // d = g*8 .. g*8+7
    qb4[2] = qp4[g * 2 + 4];  qb4[3] = qp4[g * 2 + 5];   // d = 16+g*8 ..
    qb4[4] = qp4[g * 2 + 8];  qb4[5] = qp4[g * 2 + 9];   // d = 32+g*8 ..
    qb4[6] = qp4[g * 2 + 12]; qb4[7] = qp4[g * 2 + 13];  // d = 48+g*8 ..
#pragma unroll
    for (int i = 0; i < 16; ++i) {
      aq0[i] = (_Float16)(qbuf[i] * 0.125f);
      aq1[i] = (_Float16)(qbuf[16 + i] * 0.125f);
    }
  }

  float m_run[8], l_run[8];
#pragma unroll
  for (int r = 0; r < 8; ++r) { m_run[r] = NEGINF; l_run[r] = 0.0f; }
  v8f o[4] = {};

  for (int kt = 0; kt < KN / 32; ++kt) {       // 50 key tiles of 32
    const int kb = kt * 32;

    // ---- stage K (row-major) and V (transposed), branchless zero-fill ----
    __syncthreads();
    {
      const int j   = kb + lane;               // lane-unique key window index
      const int tw  = j >> 5;                  // /N
      const int tkk = c * CSZ - TWIN + tw;
      const int inr = (int)(tkk >= 0) & (int)(tkk < BT);
      const size_t tok = inr ? (size_t)(tkk * NN + (j & 31)) : 0;
      const float vf = (float)(inr & (int)(pm[tok] != 0));   // 0.0 or 1.0
      const float4* kp4 = (const float4*)(qkv + tok * (3 * DD) + DD + head * HD);
      const float4* vp4 = (const float4*)(qkv + tok * (3 * DD) + 2 * DD + head * HD);
#pragma unroll
      for (int q4 = 0; q4 < 16; ++q4) {
        const float4 kk = kp4[q4];
        const float4 vv = vp4[q4];
        const int d = q4 * 4;
        Kls[lane * 72 + d + 0] = (_Float16)(kk.x * vf);
        Kls[lane * 72 + d + 1] = (_Float16)(kk.y * vf);
        Kls[lane * 72 + d + 2] = (_Float16)(kk.z * vf);
        Kls[lane * 72 + d + 3] = (_Float16)(kk.w * vf);
        Vt[(d + 0) * 40 + lane] = (_Float16)(vv.x * vf);
        Vt[(d + 1) * 40 + lane] = (_Float16)(vv.y * vf);
        Vt[(d + 2) * 40 + lane] = (_Float16)(vv.z * vf);
        Vt[(d + 3) * 40 + lane] = (_Float16)(vv.w * vf);
      }
    }
    __syncthreads();

    // ---- per-lane key-column metadata (branchless) ----
    int tkw[2], nk[2], kval[2];
#pragma unroll
    for (int ct = 0; ct < 2; ++ct) {
      const int j = kb + ct * 16 + lr;
      tkw[ct] = j >> 5;
      nk[ct]  = j & 31;
      const int tk = c * CSZ - TWIN + tkw[ct];
      const int inr = (int)(tk >= 0) & (int)(tk < BT);
      const size_t tok = inr ? (size_t)(tk * NN + nk[ct]) : 0;
      kval[ct] = inr & (int)(pm[tok] != 0);
    }

    // ---- K^T fragments from LDS + S = Q K^T (4 WMMAs) ----
    v8f s[2] = {};
#pragma unroll
    for (int ct = 0; ct < 2; ++ct) {
      v16h bk0, bk1;
      const int kbase = (ct * 16 + lr) * 72;
#pragma unroll
      for (int i = 0; i < 8; ++i) {
        bk0[i]     = Kls[kbase + g * 8 + i];
        bk0[i + 8] = Kls[kbase + 16 + g * 8 + i];
        bk1[i]     = Kls[kbase + 32 + g * 8 + i];
        bk1[i + 8] = Kls[kbase + 48 + g * 8 + i];
      }
      s[ct] = wmma_f16(aq0, bk0, s[ct]);
      s[ct] = wmma_f16(aq1, bk1, s[ct]);
    }

    // ---- mask + online softmax (rows M = r + 8*g live on 16 lanes of group g)
    float p0[8], p1[8], alpha[8];
#pragma unroll
    for (int r = 0; r < 8; ++r) {
      const int qi = qbase + r + 8 * g;
      const int tq = qi >> 5, nq = qi & 31;
      const int td0 = abs(tq + TWIN - tkw[0]);
      const int td1 = abs(tq + TWIN - tkw[1]);
      const bool m0b = kval[0] && ((td0 == 0) | ((nq == nk[0]) & (td0 <= TWIN)));
      const bool m1b = kval[1] && ((td1 == 0) | ((nq == nk[1]) & (td1 <= TWIN)));
      const float sv0 = m0b ? s[0][r] : NEGINF;
      const float sv1 = m1b ? s[1][r] : NEGINF;
      float tmax = fmaxf(sv0, sv1);
#pragma unroll
      for (int off = 8; off >= 1; off >>= 1)
        tmax = fmaxf(tmax, __shfl_xor(tmax, off, 32));
      const float mn = fmaxf(m_run[r], tmax);
      const float al = (m_run[r] == NEGINF) ? 0.0f : __expf(m_run[r] - mn);
      const float e0 = m0b ? __expf(sv0 - mn) : 0.0f;
      const float e1 = m1b ? __expf(sv1 - mn) : 0.0f;
      float rs = e0 + e1;
#pragma unroll
      for (int off = 8; off >= 1; off >>= 1)
        rs += __shfl_xor(rs, off, 32);
      l_run[r] = l_run[r] * al + rs;
      m_run[r] = mn;
      p0[r] = e0; p1[r] = e1; alpha[r] = al;
    }
#pragma unroll
    for (int t = 0; t < 4; ++t)
#pragma unroll
      for (int r = 0; r < 8; ++r) o[t][r] *= alpha[r];

    // ---- transpose P (C layout -> A fragment) through LDS ----
    __syncthreads();
#pragma unroll
    for (int r = 0; r < 8; ++r) {
      Pls[r + 8 * g][lr]      = (_Float16)p0[r];
      Pls[r + 8 * g][16 + lr] = (_Float16)p1[r];
    }
    __syncthreads();
    v16h ap;
#pragma unroll
    for (int i = 0; i < 8; ++i) {
      ap[i]     = Pls[lr][g * 8 + i];
      ap[i + 8] = Pls[lr][16 + g * 8 + i];
    }

    // ---- O += P V : 4 WMMAs over 4 d-tiles, V fragments from Vt (b128) ----
#pragma unroll
    for (int t = 0; t < 4; ++t) {
      v16h bv;
      const int vbase = (t * 16 + lr) * 40;
#pragma unroll
      for (int i = 0; i < 8; ++i) {
        bv[i]     = Vt[vbase + g * 8 + i];
        bv[i + 8] = Vt[vbase + 16 + g * 8 + i];
      }
      o[t] = wmma_f16(ap, bv, o[t]);
    }
  }

  // ---- normalize and store (rows masked to zero when fully masked) ----
#pragma unroll
  for (int t = 0; t < 4; ++t)
#pragma unroll
    for (int r = 0; r < 8; ++r) {
      const float l = l_run[r];
      const float val = (l > 0.0f) ? (o[t][r] / l) : 0.0f;
      const int tok = c * QN + qbase + r + 8 * g;
      out[(size_t)tok * DD + head * HD + t * 16 + lr] = val;
    }
}

// ---------------------------------------------------------------------------
// Fused residual + (optional person-mask on the added branch) + LayerNorm.
// One 128-thread block per 512-wide row; 4 consecutive elems/thread (b128).
// out = LN(h + mask*add)*gamma + beta
// ---------------------------------------------------------------------------
__global__ __launch_bounds__(128)
void resid_layernorm(const float* __restrict__ h, const float* __restrict__ add,
                     const unsigned char* __restrict__ pm,
                     const float* __restrict__ gamma, const float* __restrict__ beta,
                     float* __restrict__ out) {
  const int row = blockIdx.x;
  const int tid = threadIdx.x;
  __shared__ float red[128];

  const float msk = pm ? (pm[row] ? 1.0f : 0.0f) : 1.0f;
  const float4 hv = ((const float4*)(h + (size_t)row * DD))[tid];
  const float4 av = ((const float4*)(add + (size_t)row * DD))[tid];

  float v[4];
  v[0] = hv.x + msk * av.x; v[1] = hv.y + msk * av.y;
  v[2] = hv.z + msk * av.z; v[3] = hv.w + msk * av.w;
  float s = v[0] + v[1] + v[2] + v[3];

  red[tid] = s; __syncthreads();
  for (int o = 64; o > 0; o >>= 1) { if (tid < o) red[tid] += red[tid + o]; __syncthreads(); }
  const float mean = red[0] * (1.0f / DD);
  __syncthreads();

  float s2 = 0.0f;
#pragma unroll
  for (int j = 0; j < 4; ++j) { const float d = v[j] - mean; s2 += d * d; }
  red[tid] = s2; __syncthreads();
  for (int o = 64; o > 0; o >>= 1) { if (tid < o) red[tid] += red[tid + o]; __syncthreads(); }
  const float var = red[0] * (1.0f / DD);
  const float inv = rsqrtf(var + 1e-5f);

  const float4 gv = ((const float4*)gamma)[tid];
  const float4 bv = ((const float4*)beta)[tid];
  float4 ov;
  ov.x = (v[0] - mean) * inv * gv.x + bv.x;
  ov.y = (v[1] - mean) * inv * gv.y + bv.y;
  ov.z = (v[2] - mean) * inv * gv.z + bv.z;
  ov.w = (v[3] - mean) * inv * gv.w + bv.w;
  ((float4*)(out + (size_t)row * DD))[tid] = ov;
}

// ---------------------------------------------------------------------------
// Launch
// ---------------------------------------------------------------------------
extern "C" void kernel_launch(void* const* d_in, const int* in_sizes, int n_in,
                              void* d_out, int out_size, void* d_ws, size_t ws_size,
                              hipStream_t stream) {
  const float*         x    = (const float*)d_in[0];
  const unsigned char* pm   = (const unsigned char*)d_in[1];   // jnp.bool_ = 1 byte
  const float*         Win  = (const float*)d_in[2];
  const float*         b_in = (const float*)d_in[3];
  const float*         Wqkv = (const float*)d_in[4];
  const float*         bqkv = (const float*)d_in[5];
  const float*         Wo   = (const float*)d_in[6];
  const float*         bo   = (const float*)d_in[7];
  const float*         g1   = (const float*)d_in[8];
  const float*         be1  = (const float*)d_in[9];
  const float*         g2   = (const float*)d_in[10];
  const float*         be2  = (const float*)d_in[11];
  const float*         Wf1  = (const float*)d_in[12];
  const float*         bf1  = (const float*)d_in[13];
  const float*         Wf2  = (const float*)d_in[14];
  const float*         bf2  = (const float*)d_in[15];
  float* out = (float*)d_out;

  // workspace layout (floats): h[TOK*512] | big[TOK*2048] | tmp[TOK*512]
  float* h   = (float*)d_ws;
  float* big = h + (size_t)TOK * DD;
  float* tmp = big + (size_t)TOK * 4 * DD;

  const dim3 blk256(256);
  const int Mtiles = TOK / 64;   // 120

  // h = x @ Win + b_in              (7680 x 256 -> 512)
  gemm_wmma<DD, DIN, 0><<<dim3(DD / 128, Mtiles), blk256, 0, stream>>>(
      x, Win, b_in, h, TOK);

  for (int l = 0; l < LL; ++l) {
    // qkv = h @ Wqkv[l] + bqkv[l]   (512 -> 1536)
    gemm_wmma<3 * DD, DD, 0><<<dim3((3 * DD) / 128, Mtiles), blk256, 0, stream>>>(
        h, Wqkv + (size_t)l * DD * 3 * DD, bqkv + (size_t)l * 3 * DD, big, TOK);

    // windowed attention -> tmp (pre-projection)
    attn_wmma<<<dim3(CC * HH * (QN / 16)), dim3(32), 0, stream>>>(big, pm, tmp);

    // proj = tmp @ Wo[l] + bo[l] -> big (reuse)
    gemm_wmma<DD, DD, 0><<<dim3(DD / 128, Mtiles), blk256, 0, stream>>>(
        tmp, Wo + (size_t)l * DD * DD, bo + (size_t)l * DD, big, TOK);

    // h = LN(h + pm*proj) * g1 + be1
    resid_layernorm<<<dim3(TOK), dim3(128), 0, stream>>>(
        h, big, pm, g1 + (size_t)l * DD, be1 + (size_t)l * DD, h);

    // ffn1 = gelu(h @ Wf1[l] + bf1[l])  (512 -> 2048) -> big
    gemm_wmma<4 * DD, DD, 1><<<dim3((4 * DD) / 128, Mtiles), blk256, 0, stream>>>(
        h, Wf1 + (size_t)l * DD * 4 * DD, bf1 + (size_t)l * 4 * DD, big, TOK);

    // ffn2 = big @ Wf2[l] + bf2[l]      (2048 -> 512) -> tmp
    gemm_wmma<DD, 4 * DD, 0><<<dim3(DD / 128, Mtiles), blk256, 0, stream>>>(
        big, Wf2 + (size_t)l * 4 * DD * DD, bf2 + (size_t)l * DD, tmp, TOK);

    // h = LN(h + ffn2) * g2 + be2   (final layer writes d_out)
    float* dst = (l == LL - 1) ? out : h;
    resid_layernorm<<<dim3(TOK), dim3(128), 0, stream>>>(
        h, tmp, nullptr, g2 + (size_t)l * DD, be2 + (size_t)l * DD, dst);
  }
}
